// Model_15676630630728
// MI455X (gfx1250) — matrix operand order repaired and checked by reading, not compiled, not run on hardware
//
#include <hip/hip_runtime.h>

typedef __attribute__((ext_vector_type(16))) __bf16 v16bf;
typedef __attribute__((ext_vector_type(8)))  __bf16 v8bf;
typedef __attribute__((ext_vector_type(8)))  float  v8f;

union V16U { v16bf v; v8bf h[2]; };
union PairU { __bf16 h[2]; unsigned u; };

#define ASTR 64   // As row stride (elements): 16 rows x 64 k
#define BSTR 72   // Bs col stride (elements): padded (144B, 16B-aligned, bank-friendly)

// ---------------------------------------------------------------------------
// zero fill
// ---------------------------------------------------------------------------
__global__ void zero_kernel(float* __restrict__ p, long n) {
    long i = (long)blockIdx.x * blockDim.x + threadIdx.x;
    if (i < n) p[i] = 0.0f;
}

// ---------------------------------------------------------------------------
// Per-column embedding: h[n, c*64+j] = x[n,c]*W[c,j] + b[c,j]  (bf16 out)
// ---------------------------------------------------------------------------
__global__ void embed_kernel(const float* __restrict__ x,
                             const float* __restrict__ W,
                             const float* __restrict__ b,
                             __bf16* __restrict__ h,
                             long total, int C) {
    long i = (long)blockIdx.x * blockDim.x + threadIdx.x;
    if (i >= total) return;
    int D = C * 64;
    long n   = i / D;
    int  rem = (int)(i - n * D);
    int  c   = rem >> 6;
    int  j   = rem & 63;
    h[i] = (__bf16)(x[n * C + c] * W[c * 64 + j] + b[c * 64 + j]);
}

// ---------------------------------------------------------------------------
// Shared tile math: stage W tile transposed, do 2 wmmas per 64-wide K step
// ---------------------------------------------------------------------------
template <typename AT>
__device__ __forceinline__ void stage_a_row(__bf16* As, const AT* __restrict__ rowp,
                                            int r, int kp) {
    // pack 2 consecutive k elements into one 32-bit LDS store
    PairU t;
    t.h[0] = (__bf16)(float)rowp[2 * kp];
    t.h[1] = (__bf16)(float)rowp[2 * kp + 1];
    *(unsigned*)&As[r * ASTR + 2 * kp] = t.u;
}

__device__ __forceinline__ void stage_b(__bf16* Bs, const float* __restrict__ W,
                                        int N, int k0, int kp, int cc) {
    PairU t;
    t.h[0] = (__bf16)W[(size_t)(k0 + 2 * kp) * N + cc];
    t.h[1] = (__bf16)W[(size_t)(k0 + 2 * kp + 1) * N + cc];
    *(unsigned*)&Bs[cc * BSTR + 2 * kp] = t.u;
}

__device__ __forceinline__ v8f tile_mma(const __bf16* As, const __bf16* Bs,
                                        int arow, int koff, int kb, int col, v8f acc) {
#pragma unroll
    for (int s = 0; s < 2; ++s) {
        const int ks = s * 32;
        V16U a, b;
        a.h[0] = *(const v8bf*)&As[arow * ASTR + ks + koff];
        a.h[1] = *(const v8bf*)&As[arow * ASTR + ks + 16 + koff];
        b.h[0] = *(const v8bf*)&Bs[col * BSTR + ks + (kb >> 1)];
        b.h[1] = *(const v8bf*)&Bs[col * BSTR + ks + (kb >> 1) + 16];
        acc = __builtin_amdgcn_wmma_f32_16x16x32_bf16(
            false, a.v, false, b.v, (short)0, acc, false, false);
    }
    return acc;
}

// ---------------------------------------------------------------------------
// Dense row-tile GEMM: out[M,N] += A[M,K] @ W[K,N] (+ bias)
// grid.x = M/16, blockDim.x = (N/16)*32, N in {16,64,128}, K % 64 == 0
// ---------------------------------------------------------------------------
template <typename AT>
__global__ void gemm_rows_kernel(const AT* __restrict__ A, int K,
                                 const float* __restrict__ W,
                                 const float* __restrict__ bias,
                                 float* __restrict__ out, int N) {
    const int tid  = threadIdx.x;
    const int nt   = blockDim.x;
    const int lane = tid & 31;
    const int wave = tid >> 5;
    const int row0 = blockIdx.x * 16;

    __shared__ __bf16 As[16 * ASTR];
    __shared__ __bf16 Bs[128 * BSTR];

    v8f acc = {};

    const int arow = lane & 15;
    const int koff = (lane < 16) ? 0 : 8;
    const int kb   = (lane < 16) ? 0 : 16;
    const int col  = (wave << 4) + (lane & 15);

    for (int k0 = 0; k0 < K; k0 += 64) {
        for (int i = tid; i < 16 * 32; i += nt) {
            int r = i >> 5, kp = i & 31;
            stage_a_row(As, &A[(size_t)(row0 + r) * K + k0], r, kp);
        }
        for (int i = tid; i < N * 32; i += nt) {
            int kp = i / N, cc = i - kp * N;
            stage_b(Bs, W, N, k0, kp, cc);
        }
        __syncthreads();
        acc = tile_mma(As, Bs, arow, koff, kb, col, acc);
        __syncthreads();
    }

    const int rbase = (lane < 16) ? 0 : 8;
#pragma unroll
    for (int j = 0; j < 8; ++j) {
        int row = row0 + rbase + j;
        float v = acc[j] + (bias ? bias[col] : 0.0f);
        out[(size_t)row * N + col] += v;
    }
}

// ---------------------------------------------------------------------------
// Fused gather -> GEMM -> scatter for SAGE aggregation:
//   out_d[dst[e], :] += h_s[src[e], :] @ Wl      (N fixed = 128)
// grid.x = E/16, blockDim.x = 256
// ---------------------------------------------------------------------------
template <typename AT>
__global__ void edge_gemm_kernel(const AT* __restrict__ hsrc,
                                 const int* __restrict__ src,
                                 const int* __restrict__ dst,
                                 int K,
                                 const float* __restrict__ Wl,
                                 float* __restrict__ outd) {
    const int tid  = threadIdx.x;
    const int lane = tid & 31;
    const int wave = tid >> 5;
    const long e0  = (long)blockIdx.x * 16;

    __shared__ int    s_src[16];
    __shared__ int    s_dst[16];
    __shared__ __bf16 As[16 * ASTR];
    __shared__ __bf16 Bs[128 * BSTR];

    if (tid < 16) {
        s_src[tid] = src[e0 + tid];
        s_dst[tid] = dst[e0 + tid];
    }
    __syncthreads();

    v8f acc = {};

    const int arow = lane & 15;
    const int koff = (lane < 16) ? 0 : 8;
    const int kb   = (lane < 16) ? 0 : 16;
    const int col  = (wave << 4) + (lane & 15);

    for (int k0 = 0; k0 < K; k0 += 64) {
        for (int i = tid; i < 16 * 32; i += 256) {
            int r = i >> 5, kp = i & 31;
            stage_a_row(As, &hsrc[(size_t)s_src[r] * K + k0], r, kp);
        }
        for (int i = tid; i < 128 * 32; i += 256) {
            int kp = i >> 7, cc = i & 127;
            stage_b(Bs, Wl, 128, k0, kp, cc);
        }
        __syncthreads();
        acc = tile_mma(As, Bs, arow, koff, kb, col, acc);
        __syncthreads();
    }

    const int rbase = (lane < 16) ? 0 : 8;
#pragma unroll
    for (int j = 0; j < 8; ++j) {
        atomicAdd(&outd[(size_t)s_dst[rbase + j] * 128 + col], acc[j]);
    }
}

// ---------------------------------------------------------------------------
// Host orchestration
// ---------------------------------------------------------------------------
extern "C" void kernel_launch(void* const* d_in, const int* in_sizes, int n_in,
                              void* d_out, int out_size, void* d_ws, size_t ws_size,
                              hipStream_t stream) {
    (void)in_sizes; (void)n_in; (void)out_size; (void)ws_size;

    // --- node tables: customer(0), order(1), product(2)
    const int  NT[3] = {50000, 200000, 20000};
    const int  C_[3] = {8, 12, 10};
    const int  D0[3] = {512, 768, 640};   // embedding widths = C*64
    // --- edge types in reference order: o2c, c2o, o2p, p2o
    const int  E_[4]   = {200000, 200000, 400000, 400000};
    const int  esrc[4] = {1, 0, 1, 2};
    const int  edst[4] = {0, 1, 2, 1};

    const float* x[3]  = {(const float*)d_in[0], (const float*)d_in[1], (const float*)d_in[2]};
    const int*   ei[4] = {(const int*)d_in[3], (const int*)d_in[4],
                          (const int*)d_in[5], (const int*)d_in[6]};
    auto P = [&](int i) { return (const float*)d_in[i]; };
    // param indices: emb W/b = 7..12; L0 edge t: Wl=13+3t, bl=14+3t, Wr=15+3t;
    // L1 edge t: 25+3t..; mlp: W0=37 b0=38 W1=39 b1=40

    // --- workspace carving ---
    char*   w = (char*)d_ws;
    size_t  off = 0;
    __bf16* h0[3];
    float*  out2[3];   // layer-1 outputs reuse (dead) h0 regions
    float*  out0[3];
    for (int i = 0; i < 3; ++i) {
        h0[i]   = (__bf16*)(w + off);
        out2[i] = (float*)(w + off);              // alias, used after h0 dies
        off    += (size_t)NT[i] * D0[i] * sizeof(__bf16);
    }
    for (int i = 0; i < 3; ++i) {
        out0[i] = (float*)(w + off);
        off    += (size_t)NT[i] * 128 * sizeof(float);
    }
    float* tmp = (float*)(w + off);               // head hidden [50000,64]
    off += (size_t)NT[0] * 64 * sizeof(float);

    auto zero = [&](float* p, long n) {
        zero_kernel<<<dim3((unsigned)((n + 255) / 256)), 256, 0, stream>>>(p, n);
    };

    // --- 1) embeddings (bf16) ---
    for (int i = 0; i < 3; ++i) {
        long total = (long)NT[i] * D0[i];
        embed_kernel<<<dim3((unsigned)((total + 255) / 256)), 256, 0, stream>>>(
            x[i], P(7 + 2 * i), P(8 + 2 * i), h0[i], total, C_[i]);
    }

    // --- 2) GNN layer 0: bf16 inputs ---
    for (int i = 0; i < 3; ++i) zero(out0[i], (long)NT[i] * 128);
    for (int t = 0; t < 4; ++t) {
        int s = esrc[t], d = edst[t];
        edge_gemm_kernel<__bf16><<<dim3(E_[t] / 16), 256, 0, stream>>>(
            h0[s], ei[t], ei[t] + E_[t], D0[s], P(13 + 3 * t), out0[d]);
        gemm_rows_kernel<__bf16><<<dim3(NT[d] / 16), 256, 0, stream>>>(
            h0[d], D0[d], P(15 + 3 * t), P(14 + 3 * t), out0[d], 128);
    }

    // --- 3) GNN layer 1: f32 inputs (out0), outputs into reused h0 regions ---
    for (int i = 0; i < 3; ++i) zero(out2[i], (long)NT[i] * 128);
    for (int t = 0; t < 4; ++t) {
        int s = esrc[t], d = edst[t];
        edge_gemm_kernel<float><<<dim3(E_[t] / 16), 256, 0, stream>>>(
            out0[s], ei[t], ei[t] + E_[t], 128, P(25 + 3 * t), out2[d]);
        gemm_rows_kernel<float><<<dim3(NT[d] / 16), 256, 0, stream>>>(
            out0[d], 128, P(27 + 3 * t), P(26 + 3 * t), out2[d], 128);
    }

    // --- 4) MLP head on customer: 128 -> 64 -> 16 ---
    zero(tmp, (long)NT[0] * 64);
    gemm_rows_kernel<float><<<dim3(NT[0] / 16), 128, 0, stream>>>(
        out2[0], 128, P(37), P(38), tmp, 64);
    zero((float*)d_out, (long)NT[0] * 16);
    gemm_rows_kernel<float><<<dim3(NT[0] / 16), 32, 0, stream>>>(
        tmp, 64, P(39), P(40), (float*)d_out, 16);
}
